// CQDBeam_43147241455655
// MI455X (gfx1250) — matrix operand order repaired and checked by reading, not compiled, not run on hardware
//
#include <hip/hip_runtime.h>
#include <hip/hip_bf16.h>

typedef __attribute__((ext_vector_type(16))) _Float16 v16h;
typedef __attribute__((ext_vector_type(8)))  float    v8f;
typedef unsigned int       u32;
typedef unsigned long long u64;

#define NTERM   128   // B
#define BEAM    128
#define DIM     64
#define NBINS   64
#define CANDCAP 4096

// ---- monotonic float <-> uint mapping (order-preserving) ----
__device__ __forceinline__ u32 f2mono(float x) {
    u32 u = __float_as_uint(x);
    return (u & 0x80000000u) ? ~u : (u | 0x80000000u);
}
__device__ __forceinline__ float mono2f(u32 u) {
    u32 b = (u & 0x80000000u) ? (u & 0x7fffffffu) : ~u;
    return __uint_as_float(b);
}

// ---- init workspace ----
__global__ void cqd_init(float* Q, float* bias, u32* minU, u32* maxU,
                         u32* candCount, u32* hist) {
    int i = blockIdx.x * blockDim.x + threadIdx.x;
    if (i < NTERM * DIM) Q[i] = 0.0f;
    if (i < NTERM) {
        bias[i] = 0.0f;
        minU[i] = 0xFFFFFFFFu;
        maxU[i] = 0u;
        candCount[i] = 0u;
    }
    if (i < NTERM * NBINS) hist[i] = 0u;
}

// ---- fold edges into per-term query vectors: Q[b] = sum sign_e * (e_emb[h]*r_emb[r]) ----
__global__ void cqd_prep(const int* __restrict__ head, const int* __restrict__ rel,
                         const int* __restrict__ neg,  const int* __restrict__ e2t,
                         const float* __restrict__ src,
                         const float* __restrict__ eemb, const float* __restrict__ remb,
                         float* Q, float* bias, int E) {
    int e = blockIdx.x;
    int d = threadIdx.x;               // blockDim = 64
    if (e >= E) return;
    int t = e2t[e];
    float sg = neg[e] ? -1.0f : 1.0f;
    float v = sg * eemb[(size_t)head[e] * DIM + d] * remb[(size_t)rel[e] * DIM + d];
    atomicAdd(&Q[t * DIM + d], v);
    if (d == 0) atomicAdd(&bias[t], src[e]);
}

// ---- WMMA scoring pass.  MODE 0: min/max, 1: histogram, 2: collect candidates ----
// One block (8 waves) stages a 16-entity x 64-dim tile in LDS once; wave w
// consumes it for M-tile w (terms [w*16, +16)), so entity_emb is read exactly
// once per pass (HBM once, then L2-resident for the later passes).
template <int MODE>
__global__ __launch_bounds__(256) void cqd_score(
    const float* __restrict__ entity, int N, int NT16,
    const float* __restrict__ Q, const float* __restrict__ bias,
    u32* minU, u32* maxU, const float* __restrict__ thresh,
    u32* hist, u32* candCount, u64* cand) {

    const int wave  = threadIdx.x >> 5;      // 0..7 == M-tile
    const int lane  = threadIdx.x & 31;
    const int hf    = lane >> 4;             // lane half
    const int ln    = lane & 15;

    __shared__ _Float16 tile[16 * DIM];                   // 2 KB staged entity tile
    __shared__ u32 lh[(MODE == 1) ? NTERM * NBINS : 1];   // 32 KB hist (MODE 1)

    // ---- A fragments: Q tile 16x64 f16, per documented 16-bit A layout ----
    // lane covers M = ln; V0..3 hold K {0..7}+8*hf, V4..7 hold K {16..23}+8*hf
    const float* qrow = Q + (size_t)(wave * 16 + ln) * DIM;
    const int khi = hf ? 8 : 0;
    v16h a0, a1;
#pragma unroll
    for (int i = 0; i < 16; ++i) {
        int v = i >> 1, p = i & 1;
        int k = (v < 4) ? (v * 2 + p + khi) : (16 + (v - 4) * 2 + p + khi);
        a0[i] = (_Float16)qrow[k];
        a1[i] = (_Float16)qrow[32 + k];
    }

    // per-register term bias (reg r covers term wave*16 + r + hf*8)
    float tb[8];
#pragma unroll
    for (int r = 0; r < 8; ++r) tb[r] = bias[wave * 16 + r + hf * 8];

    float tmin[8], tsc[8], tth[8];
    if (MODE == 1) {
#pragma unroll
        for (int r = 0; r < 8; ++r) {
            int term = wave * 16 + r + hf * 8;
            float mn = mono2f(minU[term]);
            float rg = mono2f(maxU[term]) - mn;
            if (!(rg > 1e-30f)) rg = 1.0f;
            tmin[r] = mn; tsc[r] = (float)NBINS / rg;
        }
        for (int i = threadIdx.x; i < NTERM * NBINS; i += 256) lh[i] = 0u;
    }
    if (MODE == 2) {
#pragma unroll
        for (int r = 0; r < 8; ++r) tth[r] = thresh[wave * 16 + r + hf * 8];
    }

    float lmin[8], lmax[8];
    if (MODE == 0) {
#pragma unroll
        for (int r = 0; r < 8; ++r) { lmin[r] = 1e38f; lmax[r] = -1e38f; }
    }

    // cooperative staging role: thread covers row rrow, cols [quad*4, +4)
    const int rrow = threadIdx.x >> 4;
    const int quad = threadIdx.x & 15;

    // block-uniform grid-stride over 16-entity subtiles
    for (int st = blockIdx.x; st < NT16; st += gridDim.x) {
        // stage: 16 rows x 64 f32 -> f16 in LDS (each thread one float4)
        int grow = st * 16 + rrow;
        if (grow >= N) grow = N - 1;
        const float4 f = *(const float4*)(entity + (size_t)grow * DIM + quad * 4);

        // prefetch next tile's chunk (global_prefetch_b8)
        {
            int pn = (st + gridDim.x) * 16 + rrow;
            if (pn < N) __builtin_prefetch(entity + (size_t)pn * DIM + quad * 4, 0, 0);
        }

        __syncthreads();   // previous iteration's consumers done
        {
            _Float16* t = &tile[rrow * DIM + quad * 4];
            t[0] = (_Float16)f.x; t[1] = (_Float16)f.y;
            t[2] = (_Float16)f.z; t[3] = (_Float16)f.w;
        }
        __syncthreads();   // tile visible to all waves

        // B fragments from LDS: lane holds column N = ln, K {0..7}+8*hf then {16..23}+8*hf
        v16h b0, b1;
        {
            const _Float16* p = &tile[ln * DIM + hf * 8];
#pragma unroll
            for (int i = 0; i < 16; ++i) {
                int kk = (i < 8) ? i : (i + 8);
                b0[i] = p[kk];
                b1[i] = p[32 + kk];
            }
        }

        v8f acc = {};
        acc = __builtin_amdgcn_wmma_f32_16x16x32_f16(false, a0, false, b0,
                                                     (short)0, acc, false, false);
        acc = __builtin_amdgcn_wmma_f32_16x16x32_f16(false, a1, false, b1,
                                                     (short)0, acc, false, false);

        int ent = st * 16 + ln;
        bool valid = ent < N;

        if (MODE == 0) {
            if (valid) {
#pragma unroll
                for (int r = 0; r < 8; ++r) {
                    float s = acc[r] + tb[r];
                    lmin[r] = fminf(lmin[r], s);
                    lmax[r] = fmaxf(lmax[r], s);
                }
            }
        } else if (MODE == 1) {
            if (valid) {
#pragma unroll
                for (int r = 0; r < 8; ++r) {
                    float s = acc[r] + tb[r];
                    int bin = (int)((s - tmin[r]) * tsc[r]);
                    bin = bin < 0 ? 0 : (bin > NBINS - 1 ? NBINS - 1 : bin);
                    atomicAdd(&lh[(wave * 16 + r + hf * 8) * NBINS + bin], 1u);
                }
            }
        } else {
            if (valid) {
#pragma unroll
                for (int r = 0; r < 8; ++r) {
                    float s = acc[r] + tb[r];
                    if (s >= tth[r]) {
                        int term = wave * 16 + r + hf * 8;
                        u32 pos = atomicAdd(&candCount[term], 1u);
                        if (pos < CANDCAP) {
                            u64 key = ((u64)f2mono(s) << 32) | (u32)(~(u32)ent);
                            cand[(size_t)term * CANDCAP + pos] = key;
                        }
                    }
                }
            }
        }
    }

    if (MODE == 0) {
        // reduce each term across the 16 lanes of its half, then one atomic
#pragma unroll
        for (int r = 0; r < 8; ++r) {
            float mn = lmin[r], mx = lmax[r];
            for (int m = 1; m < 16; m <<= 1) {
                mn = fminf(mn, __shfl_xor(mn, m, 32));
                mx = fmaxf(mx, __shfl_xor(mx, m, 32));
            }
            if (ln == 0) {
                int term = wave * 16 + r + hf * 8;
                atomicMin(&minU[term], f2mono(mn));
                atomicMax(&maxU[term], f2mono(mx));
            }
        }
    } else if (MODE == 1) {
        __syncthreads();
        for (int i = threadIdx.x; i < NTERM * NBINS; i += 256) {
            u32 c = lh[i];
            if (c) atomicAdd(&hist[i], c);
        }
    }
}

// ---- per-term cutoff: highest bin s.t. count(score >= binStart) >= BEAM ----
__global__ void cqd_scan(const u32* __restrict__ minU, const u32* __restrict__ maxU,
                         const u32* __restrict__ hist, float* thresh) {
    int b = threadIdx.x;
    if (b >= NTERM) return;
    float mn = mono2f(minU[b]);
    float rg = mono2f(maxU[b]) - mn;
    if (!(rg > 1e-30f)) { thresh[b] = mn; return; }
    const u32* h = hist + (size_t)b * NBINS;
    u32 cum = 0; int bin = 0;
    for (int i = NBINS - 1; i >= 0; --i) {
        cum += h[i];
        if (cum >= BEAM) { bin = i; break; }
    }
    thresh[b] = mn + (rg / (float)NBINS) * (float)bin;
}

// ---- exact top-128 per term: bitonic sort of candidates in LDS ----
__global__ __launch_bounds__(256) void cqd_topk(const u64* __restrict__ cand,
                                                const u32* __restrict__ candCount,
                                                float* __restrict__ out) {
    __shared__ u64 buf[CANDCAP];    // 32 KB
    int b = blockIdx.x;
    u32 cnt = candCount[b]; if (cnt > CANDCAP) cnt = CANDCAP;
    const u64* src = cand + (size_t)b * CANDCAP;
    for (int i = threadIdx.x; i < CANDCAP; i += 256)
        buf[i] = (i < (int)cnt) ? src[i] : 0ULL;
    __syncthreads();
    for (int k = 2; k <= CANDCAP; k <<= 1) {
        for (int j = k >> 1; j > 0; j >>= 1) {
            for (int i = threadIdx.x; i < CANDCAP; i += 256) {
                int l = i ^ j;
                if (l > i) {
                    u64 x = buf[i], y = buf[l];
                    bool asc = (i & k) == 0;
                    if ((x > y) == asc) { buf[i] = y; buf[l] = x; }
                }
            }
            __syncthreads();
        }
    }
    // ascending sorted; top-BEAM are the last elements, descending
    for (int j = threadIdx.x; j < BEAM; j += 256) {
        u64 key = buf[CANDCAP - 1 - j];
        out[b * BEAM + j] = mono2f((u32)(key >> 32));
        out[NTERM * BEAM + b * BEAM + j] = (float)(~(u32)(key & 0xFFFFFFFFu));
    }
}

extern "C" void kernel_launch(void* const* d_in, const int* in_sizes, int n_in,
                              void* d_out, int out_size, void* d_ws, size_t ws_size,
                              hipStream_t stream) {
    const int*   head = (const int*)d_in[0];
    const int*   rel  = (const int*)d_in[1];
    const int*   neg  = (const int*)d_in[2];
    const int*   e2t  = (const int*)d_in[3];
    const float* src  = (const float*)d_in[4];
    const float* eemb = (const float*)d_in[5];
    const float* remb = (const float*)d_in[6];
    const int E = in_sizes[0];
    const int N = in_sizes[5] / DIM;
    const int NT16 = (N + 15) / 16;

    // workspace carve (offsets all 8B aligned)
    char* w = (char*)d_ws;
    float* Q         = (float*)(w);                 // 128*64*4 = 32768
    float* bias      = (float*)(w + 32768);         // 512
    u32*   minU      = (u32*)  (w + 33280);         // 512
    u32*   maxU      = (u32*)  (w + 33792);         // 512
    float* thresh    = (float*)(w + 34304);         // 512
    u32*   candCount = (u32*)  (w + 34816);         // 512
    u32*   hist      = (u32*)  (w + 35328);         // 128*64*4 = 32768
    u64*   cand      = (u64*)  (w + 68096);         // 128*4096*8 = 4 MB

    float* out = (float*)d_out;

    cqd_init<<<256, 256, 0, stream>>>(Q, bias, minU, maxU, candCount, hist);
    cqd_prep<<<E, DIM, 0, stream>>>(head, rel, neg, e2t, src, eemb, remb, Q, bias, E);

    const int sgrid = 512;   // block-uniform grid-stride over 12500 subtiles
    cqd_score<0><<<sgrid, 256, 0, stream>>>(eemb, N, NT16, Q, bias, minU, maxU,
                                            thresh, hist, candCount, cand);
    cqd_score<1><<<sgrid, 256, 0, stream>>>(eemb, N, NT16, Q, bias, minU, maxU,
                                            thresh, hist, candCount, cand);
    cqd_scan<<<1, NTERM, 0, stream>>>(minU, maxU, hist, thresh);
    cqd_score<2><<<sgrid, 256, 0, stream>>>(eemb, N, NT16, Q, bias, minU, maxU,
                                            thresh, hist, candCount, cand);
    cqd_topk<<<NTERM, 256, 0, stream>>>(cand, candCount, out);
}
